// Attn_30820685316537
// MI455X (gfx1250) — matrix operand order repaired and checked by reading, not compiled, not run on hardware
//
#include <hip/hip_runtime.h>
#include <hip/hip_bf16.h>

// Problem dims (fixed by the reference)
#define BB   16
#define QQ   512
#define SSS  1024
#define DDD  1024
#define NSEG 64

typedef __attribute__((ext_vector_type(16))) __bf16 v16bf;
typedef __attribute__((ext_vector_type(8)))  float  v8f;

union Frag16 {
    v16bf v;
    unsigned int u[8];
    uint4 q[2];
};

#if __has_builtin(__builtin_amdgcn_global_load_async_to_lds_b128)
#define HAVE_ASYNC_LDS 1
#define GLOBAL_AS __attribute__((address_space(1)))
#define LDS_AS    __attribute__((address_space(3)))
typedef int v4i __attribute__((vector_size(16)));
#else
#define HAVE_ASYNC_LDS 0
#endif

__device__ __forceinline__ unsigned short f2bf(float f) {
    unsigned int u = __float_as_uint(f);
    u += 0x7FFFu + ((u >> 16) & 1u);   // round-to-nearest-even
    return (unsigned short)(u >> 16);
}
__device__ __forceinline__ unsigned int pk2(unsigned short lo, unsigned short hi) {
    return (unsigned int)lo | ((unsigned int)hi << 16);
}

// 16-byte global -> LDS copy: async (ASYNCcnt) when available, else sync.
__device__ __forceinline__ void cp16(const unsigned short* g, unsigned short* l) {
#if HAVE_ASYNC_LDS
    GLOBAL_AS v4i* gp =
        (GLOBAL_AS v4i*)(GLOBAL_AS void*)(void*)const_cast<unsigned short*>(g);
    LDS_AS v4i* lp = (LDS_AS v4i*)(LDS_AS void*)(void*)l;
    __builtin_amdgcn_global_load_async_to_lds_b128(gp, lp, 0, 0);
#else
    *(uint4*)l = *(const uint4*)g;
#endif
}
__device__ __forceinline__ void wait_async_0() {
#if HAVE_ASYNC_LDS
#if __has_builtin(__builtin_amdgcn_s_wait_asynccnt)
    __builtin_amdgcn_s_wait_asynccnt(0);
#else
    asm volatile("s_wait_asynccnt 0x0" ::: "memory");
#endif
#endif
}
__device__ __forceinline__ void wait_async_2() {
#if HAVE_ASYNC_LDS
#if __has_builtin(__builtin_amdgcn_s_wait_asynccnt)
    __builtin_amdgcn_s_wait_asynccnt(2);
#else
    asm volatile("s_wait_asynccnt 0x2" ::: "memory");
#endif
#endif
}

// ---------------------------------------------------------------------------
// fp32 -> bf16 converters (bandwidth-bound; one pass each over the big inputs)
// ---------------------------------------------------------------------------
__global__ void cvt_f32_bf16(const float* __restrict__ src,
                             unsigned short* __restrict__ dst, int n) {
    int i = blockIdx.x * 256 + threadIdx.x;
    if (i < n) dst[i] = f2bf(src[i]);
}

// output -> outB  and into the right half of combined [B,Q,2D]
__global__ void cvt_output(const float* __restrict__ out,
                           unsigned short* __restrict__ outB,
                           unsigned short* __restrict__ combB) {
    int i = blockIdx.x * 256 + threadIdx.x;
    if (i < BB * QQ * DDD) {
        unsigned short v = f2bf(out[i]);
        outB[i] = v;
        int row = i / DDD, d = i - row * DDD;
        combB[(size_t)row * (2 * DDD) + DDD + d] = v;
    }
}

// ---------------------------------------------------------------------------
// Segment counts per batch (LDS atomics)
// ---------------------------------------------------------------------------
__global__ void seg_counts(const int* __restrict__ sids, int* __restrict__ counts) {
    __shared__ int c[NSEG];
    int b = blockIdx.x;
    if (threadIdx.x < NSEG) c[threadIdx.x] = 0;
    __syncthreads();
    for (int s = threadIdx.x; s < SSS; s += 256)
        atomicAdd(&c[sids[b * SSS + s]], 1);
    __syncthreads();
    if (threadIdx.x < NSEG) counts[b * NSEG + threadIdx.x] = c[threadIdx.x];
}

// ---------------------------------------------------------------------------
// Per-segment context sums: ctxsegB[b][n][d] = sum_{s in seg n} context[b][s][d]
// Exploits sorted segment_ids (contiguous segments): one running sum.
// ---------------------------------------------------------------------------
__global__ void ctx_segsum(const float* __restrict__ ctx,
                           const int* __restrict__ sids,
                           unsigned short* __restrict__ ctxsegB) {
    __shared__ int sid[SSS];
    int b = blockIdx.y;
    int d = blockIdx.x * 256 + threadIdx.x;
    for (int s = threadIdx.x; s < SSS; s += 256) sid[s] = sids[b * SSS + s];
    __syncthreads();
    const float* cb = ctx + (size_t)b * SSS * DDD + d;
    unsigned short* ob = ctxsegB + (size_t)b * NSEG * DDD + d;
    int cur = sid[0];
    for (int n = 0; n < cur; ++n) ob[(size_t)n * DDD] = 0;   // empty leading segs
    float acc = 0.f;
    for (int s = 0; s < SSS; ++s) {
        int id = sid[s];
        if (id != cur) {
            ob[(size_t)cur * DDD] = f2bf(acc);
            for (int n = cur + 1; n < id; ++n) ob[(size_t)n * DDD] = 0;
            cur = id; acc = 0.f;
        }
        acc += cb[(size_t)s * DDD];
    }
    ob[(size_t)cur * DDD] = f2bf(acc);
    for (int n = cur + 1; n < NSEG; ++n) ob[(size_t)n * DDD] = 0;
}

// ---------------------------------------------------------------------------
// Fused segment-average + softmax.
// p[n] = exp(avg[n]-m) / sum_n count[n]*exp(avg[n]-m).
// Emits full align [B,Q,S] and compact p bf16 [B,Q,64] for the K=64 GEMM.
// ---------------------------------------------------------------------------
__global__ void seg_softmax(const float* __restrict__ score,
                            const int* __restrict__ sids,
                            const int* __restrict__ counts,
                            float* __restrict__ alignOut,
                            unsigned short* __restrict__ pnB) {
    __shared__ int   sid[SSS];
    __shared__ float segsum[NSEG];
    __shared__ float pbuf[NSEG];
    __shared__ float invden;
    int bq = blockIdx.x;            // b*Q + q
    int b  = bq / QQ;
    int t  = threadIdx.x;
    for (int s = t; s < SSS; s += 256) sid[s] = sids[b * SSS + s];
    if (t < NSEG) segsum[t] = 0.f;
    __syncthreads();
    const float* sc = score + (size_t)bq * SSS;
#pragma unroll
    for (int i = 0; i < 4; ++i) {
        int s = t + 256 * i;
        atomicAdd(&segsum[sid[s]], sc[s]);      // ds_add_f32
    }
    __syncthreads();
    if (t == 0) {
        float m = -3.4e38f;
        for (int n = 0; n < NSEG; ++n) {
            int c = counts[b * NSEG + n];
            float a = segsum[n] / fmaxf((float)c, 1.f);
            pbuf[n] = a;
            if (c > 0 && a > m) m = a;
        }
        float den = 0.f;
        for (int n = 0; n < NSEG; ++n) {
            int c = counts[b * NSEG + n];
            float e = (c > 0) ? __expf(pbuf[n] - m) : 0.f;
            pbuf[n] = e;
            den += (float)c * e;
        }
        invden = 1.f / den;
    }
    __syncthreads();
    if (t < NSEG) {
        float p = pbuf[t] * invden;
        pbuf[t] = p;
        pnB[(size_t)bq * NSEG + t] = f2bf(p);
    }
    __syncthreads();
    float* ao = alignOut + (size_t)bq * SSS;
#pragma unroll
    for (int i = 0; i < 4; ++i) {
        int s = t + 256 * i;
        ao[s] = pbuf[sid[s]];
    }
}

// ---------------------------------------------------------------------------
// GEMM (TN): C[M,N] = A[M,K] (row-major) * B[N,K]^T (row-major), bf16 in /
// fp32 out, via V_WMMA_F32_16X16X32_BF16.  64x64 macro-tile, 8 waves, each
// wave owns a 32x16 strip (2 accumulators).  Double-buffered LDS with async
// global->LDS copies (ASYNCcnt); steady-state loop is branch-free (always
// prefetches the next tile) with a peeled epilogue for the final tile.
// Requires K >= 64 (true for all call sites: K = 1024 / 2048).
// Optional fused bias+tanh epilogue.
// ---------------------------------------------------------------------------
__launch_bounds__(256)
__global__ void gemm_tn_bf16(const unsigned short* __restrict__ A,
                             const unsigned short* __restrict__ Bm,
                             float* __restrict__ C,
                             const float* __restrict__ bias,
                             int K, int lda, int ldb, int ldc,
                             long long strideA, long long strideB, long long strideC,
                             int doTanh) {
    __shared__ __align__(16) unsigned short Al[2][64][40];  // 80B rows: 16B aligned
    __shared__ __align__(16) unsigned short Bl[2][64][40];
    int b = blockIdx.z;
    const unsigned short* Ab = A + (size_t)b * strideA;
    const unsigned short* Bb = Bm + (size_t)b * strideB;
    float* Cb = C + (size_t)b * strideC;
    int gm0 = blockIdx.y * 64;
    int gn0 = blockIdx.x * 64;
    int t    = threadIdx.x;
    int lane = t & 31;
    int wave = t >> 5;
    int wm = wave >> 2;          // 0..1 : M strip
    int wn = wave & 3;           // 0..3 : N strip
    int half = lane >> 4;        // lane group
    int srow = t >> 2;           // staging: row 0..63
    int scol = (t & 3) * 8;      // staging: col 0,8,16,24

    const unsigned short* aRow = Ab + (size_t)(gm0 + srow) * lda + scol;
    const unsigned short* bRow = Bb + (size_t)(gn0 + srow) * ldb + scol;
    unsigned short* la0 = &Al[0][srow][scol];
    unsigned short* la1 = &Al[1][srow][scol];
    unsigned short* lb0 = &Bl[0][srow][scol];
    unsigned short* lb1 = &Bl[1][srow][scol];

    v8f acc0 = {}; v8f acc1 = {};
    int nT = K >> 5;             // k-steps of 32 (>= 2)

    int kbA = half * 8;
    int rA  = wm * 32 + (lane & 15);
    int nB  = wn * 16 + (lane & 15);
    int khB = half * 8;

    auto compute = [&](int cur) {
        // A fragment (16x32 bf16, ISA 7.12.2): lanes 0-15 K={0..7,16..23},
        // lanes 16-31 K={8..15,24..31}; pairs packed per VGPR.
        Frag16 fa0, fa1, fb;
        fa0.q[0] = *(const uint4*)&Al[cur][rA][kbA];
        fa0.q[1] = *(const uint4*)&Al[cur][rA][16 + kbA];
        fa1.q[0] = *(const uint4*)&Al[cur][rA + 16][kbA];
        fa1.q[1] = *(const uint4*)&Al[cur][rA + 16][16 + kbA];
        // B fragment (32x16): N = lane%16, lanes 0-15 K={0..7,16..23}, 16-31 K={8..15,24..31}.
        fb.q[0] = *(const uint4*)&Bl[cur][nB][khB];
        fb.q[1] = *(const uint4*)&Bl[cur][nB][khB + 16];
        acc0 = __builtin_amdgcn_wmma_f32_16x16x32_bf16(false, fa0.v, false, fb.v,
                                                       (short)0, acc0, false, false);
        acc1 = __builtin_amdgcn_wmma_f32_16x16x32_bf16(false, fa1.v, false, fb.v,
                                                       (short)0, acc1, false, false);
    };

    // prologue: tile 0 in flight
    cp16(aRow, la0);
    cp16(bRow, lb0);

    // steady state: always prefetch tile i+1 into the other buffer
    for (int i = 0; i < nT - 1; ++i) {
        int cur = i & 1;
        cp16(aRow + (size_t)(i + 1) * 32, cur ? la0 : la1);
        cp16(bRow + (size_t)(i + 1) * 32, cur ? lb0 : lb1);
        wait_async_2();          // current tile landed; next tile's 2 in flight
        __syncthreads();         // all waves' current tile visible in LDS
        compute(cur);
        __syncthreads();         // reads done; buffer may be refilled next iter
    }

    // epilogue: last tile
    wait_async_0();
    __syncthreads();
    compute((nT - 1) & 1);

    // C layout: VGPR r -> row r + 8*half, col = lane%16
    int cn  = gn0 + wn * 16 + (lane & 15);
    int cm0 = gm0 + wm * 32 + half * 8;
    float bv = doTanh ? bias[cn] : 0.f;
#pragma unroll
    for (int r = 0; r < 8; ++r) {
        float x0 = acc0[r], x1 = acc1[r];
        if (doTanh) { x0 = tanhf(x0 + bv); x1 = tanhf(x1 + bv); }
        Cb[(size_t)(cm0 + r) * ldc + cn]      = x0;
        Cb[(size_t)(cm0 + 16 + r) * ldc + cn] = x1;
    }
}

// ---------------------------------------------------------------------------
// GEMM (NN, small-K): C[M,N] = A[M,K] * B[K,N], bf16 in, bf16 out
// (writes into the left half of the combined buffer).  K = 64 here: tiny,
// keep simple synchronous staging.
// ---------------------------------------------------------------------------
__launch_bounds__(256)
__global__ void gemm_nn_bf16out(const unsigned short* __restrict__ A,
                                const unsigned short* __restrict__ Bm,
                                unsigned short* __restrict__ C16,
                                int K, int lda, int ldb, int ldc,
                                long long strideA, long long strideB, long long strideC) {
    __shared__ __align__(16) unsigned short Al[64][40];
    __shared__ __align__(16) unsigned short Bl[32][72];   // [K][N] tile, 144B rows
    int b = blockIdx.z;
    const unsigned short* Ab = A + (size_t)b * strideA;
    const unsigned short* Bb = Bm + (size_t)b * strideB;
    unsigned short* Cb = C16 + (size_t)b * strideC;
    int gm0 = blockIdx.y * 64;
    int gn0 = blockIdx.x * 64;
    int t    = threadIdx.x;
    int lane = t & 31;
    int wave = t >> 5;
    int wm = wave >> 2;
    int wn = wave & 3;
    int half = lane >> 4;
    int arow = t >> 2;  int acol = (t & 3) * 8;     // A stage: 64x32
    int brow = t >> 3;  int bcol = (t & 7) * 8;     // B stage: 32x64
    v8f acc0 = {}; v8f acc1 = {};
    for (int k0 = 0; k0 < K; k0 += 32) {
        *(uint4*)&Al[arow][acol] =
            *(const uint4*)(Ab + (size_t)(gm0 + arow) * lda + k0 + acol);
        *(uint4*)&Bl[brow][bcol] =
            *(const uint4*)(Bb + (size_t)(k0 + brow) * ldb + gn0 + bcol);
        __syncthreads();
        int kbA = half * 8;
        Frag16 fa0, fa1, fb;
        int r = wm * 32 + (lane & 15);
        fa0.q[0] = *(const uint4*)&Al[r][kbA];
        fa0.q[1] = *(const uint4*)&Al[r][16 + kbA];
        fa1.q[0] = *(const uint4*)&Al[r + 16][kbA];
        fa1.q[1] = *(const uint4*)&Al[r + 16][16 + kbA];
        int n  = wn * 16 + (lane & 15);
        int kh = half * 8;
#pragma unroll
        for (int v = 0; v < 8; ++v)
            fb.u[v] = pk2(Bl[kh + 2 * v + (v >= 4 ? 8 : 0)][n], Bl[kh + 2 * v + 1 + (v >= 4 ? 8 : 0)][n]);
        acc0 = __builtin_amdgcn_wmma_f32_16x16x32_bf16(false, fa0.v, false, fb.v,
                                                       (short)0, acc0, false, false);
        acc1 = __builtin_amdgcn_wmma_f32_16x16x32_bf16(false, fa1.v, false, fb.v,
                                                       (short)0, acc1, false, false);
        __syncthreads();
    }
    int cn  = gn0 + wn * 16 + (lane & 15);
    int cm0 = gm0 + wm * 32 + half * 8;
#pragma unroll
    for (int r = 0; r < 8; ++r) {
        Cb[(size_t)(cm0 + r) * ldc + cn]      = f2bf(acc0[r]);
        Cb[(size_t)(cm0 + 16 + r) * ldc + cn] = f2bf(acc1[r]);
    }
}

// ---------------------------------------------------------------------------
extern "C" void kernel_launch(void* const* d_in, const int* in_sizes, int n_in,
                              void* d_out, int out_size, void* d_ws, size_t ws_size,
                              hipStream_t stream) {
    (void)in_sizes; (void)n_in; (void)out_size; (void)ws_size;
    const float* output  = (const float*)d_in[0];   // [B,Q,D]
    const float* context = (const float*)d_in[1];   // [B,S,D]
    const float* Ww      = (const float*)d_in[2];   // [D,2D]
    const float* Wb      = (const float*)d_in[3];   // [D]
    const int*   sids    = (const int*)d_in[4];     // [B,S]

    float* out      = (float*)d_out;                     // [B,Q,D]
    float* alignOut = out + (size_t)BB * QQ * DDD;       // [B,Q,S]

    // workspace carve-up (~119 MB)
    char* p = (char*)d_ws;
    auto take = [&](size_t bytes) { char* r = p; p += (bytes + 255) & ~(size_t)255; return r; };
    unsigned short* outB    = (unsigned short*)take((size_t)BB * QQ * DDD * 2);
    unsigned short* ctxB    = (unsigned short*)take((size_t)BB * SSS * DDD * 2);
    unsigned short* wB      = (unsigned short*)take((size_t)DDD * 2 * DDD * 2);
    unsigned short* combB   = (unsigned short*)take((size_t)BB * QQ * 2 * DDD * 2);
    unsigned short* pnB     = (unsigned short*)take((size_t)BB * QQ * NSEG * 2);
    unsigned short* ctxsegB = (unsigned short*)take((size_t)BB * NSEG * DDD * 2);
    float*          score   = (float*)take((size_t)BB * QQ * SSS * 4);
    int*            counts  = (int*)take((size_t)BB * NSEG * 4);

    // 1) convert inputs to bf16 (output also fills right half of combined)
    cvt_output<<<(BB * QQ * DDD + 255) / 256, 256, 0, stream>>>(output, outB, combB);
    cvt_f32_bf16<<<(BB * SSS * DDD + 255) / 256, 256, 0, stream>>>(context, ctxB, BB * SSS * DDD);
    cvt_f32_bf16<<<(DDD * 2 * DDD + 255) / 256, 256, 0, stream>>>(Ww, wB, DDD * 2 * DDD);

    // 2) segment metadata
    seg_counts<<<BB, 256, 0, stream>>>(sids, counts);
    ctx_segsum<<<dim3(DDD / 256, BB), 256, 0, stream>>>(context, sids, ctxsegB);

    // 3) score = output @ context^T  (M=Q, N=S, K=D)
    gemm_tn_bf16<<<dim3(SSS / 64, QQ / 64, BB), 256, 0, stream>>>(
        outB, ctxB, score, nullptr, DDD, DDD, DDD, SSS,
        (long long)QQ * DDD, (long long)SSS * DDD, (long long)QQ * SSS, 0);

    // 4) segment-avg + softmax -> align output + compact p[B,Q,64]
    seg_softmax<<<BB * QQ, 256, 0, stream>>>(score, sids, counts, alignOut, pnB);

    // 5) aligned_context = p @ ctx_segsum  (M=Q, N=D, K=64) -> left half of combined
    gemm_nn_bf16out<<<dim3(DDD / 64, QQ / 64, BB), 256, 0, stream>>>(
        pnB, ctxsegB, combB, NSEG, NSEG, DDD, 2 * DDD,
        (long long)QQ * NSEG, (long long)NSEG * DDD, (long long)QQ * 2 * DDD);

    // 6) out = tanh(combined @ W^T + bias)  (M=Q, N=D, K=2D)
    gemm_tn_bf16<<<dim3(DDD / 64, QQ / 64, BB), 256, 0, stream>>>(
        combB, wB, out, Wb, 2 * DDD, 2 * DDD, 2 * DDD, DDD,
        (long long)QQ * 2 * DDD, 0LL, (long long)QQ * DDD, 1);
}
